// BiRNN_C2_15882789061054
// MI455X (gfx1250) — matrix operand order repaired and checked by reading, not compiled, not run on hardware
//
#include <hip/hip_runtime.h>

typedef __bf16 bf16_t;
typedef __attribute__((ext_vector_type(16))) __bf16        v16bf;
typedef __attribute__((ext_vector_type(8)))  __bf16        v8bf;
typedef __attribute__((ext_vector_type(8)))  float         v8f;
typedef __attribute__((ext_vector_type(8)))  unsigned int  v8u;

#define B_      16
#define T_      768
#define FEAT_   76
#define K0P     96          // FEAT padded to multiple of 32
#define UNITS_  1024
#define NG      4096        // 4*UNITS gate columns
#define ROWS    (B_ * T_)   // 12288
#define NWG_DIR 32          // workgroups per direction in recurrent kernel

// ---------------------------------------------------------------------------
// fragment load helpers
// ---------------------------------------------------------------------------
// B fragment: 16 contiguous bf16 per lane (32B) -> two b128 loads
__device__ __forceinline__ v16bf ld_bfrag(const bf16_t* p) {
    uint4 lo = *(const uint4*)(p);
    uint4 hi = *(const uint4*)(p + 8);
    v8u u = {lo.x, lo.y, lo.z, lo.w, hi.x, hi.y, hi.z, hi.w};
    return __builtin_bit_cast(v16bf, u);
}
// A fragment: two 8-element runs at +0 and +16 (K-layout of 16-bit A matrix)
__device__ __forceinline__ v16bf ld_afrag(const bf16_t* p) {
    uint4 lo = *(const uint4*)(p);
    uint4 hi = *(const uint4*)(p + 16);
    v8u u = {lo.x, lo.y, lo.z, lo.w, hi.x, hi.y, hi.z, hi.w};
    return __builtin_bit_cast(v16bf, u);
}

__device__ __forceinline__ float sigf(float x) {
    return 1.0f / (1.0f + __expf(-x));
}

// ---------------------------------------------------------------------------
// init: zero h-state ping-pong buffers and grid barriers (every call, so graph
// replays after the 0xAA poison are deterministic)
// ---------------------------------------------------------------------------
__global__ void init_kernel(bf16_t* hstate, unsigned* bar) {
    int i = threadIdx.x;
    for (int j = i; j < 2 * 2 * 16 * UNITS_; j += 256) hstate[j] = (bf16_t)0.0f;
    if (i < 8) bar[i] = 0u;
}

// ---------------------------------------------------------------------------
// convert x [B,T,76] f32 -> padded bf16 [ROWS,96]
// ---------------------------------------------------------------------------
__global__ void convx_kernel(const float* __restrict__ x, bf16_t* __restrict__ xbf) {
    int idx = blockIdx.x * 256 + threadIdx.x;
    if (idx >= ROWS * K0P) return;
    int k = idx % K0P, row = idx / K0P;
    xbf[idx] = (bf16_t)(k < FEAT_ ? x[(size_t)row * FEAT_ + k] : 0.0f);
}

// ---------------------------------------------------------------------------
// pack f32 weight [Ksrc, 4096] -> bf16 WMMA-B fragment layout
// dst[nt][kt][lane][e]  with  K = kt*32 + (e>>3)*16 + (lane>>4)*8 + (e&7),  N = nt*16 + (lane&15)
// ---------------------------------------------------------------------------
__global__ void pack_b_kernel(const float* __restrict__ src, bf16_t* __restrict__ dst,
                              int Ksrc, int KT, long total) {
    long idx = (long)blockIdx.x * 256 + threadIdx.x;
    if (idx >= total) return;
    int  e    = (int)(idx & 15);
    int  lane = (int)((idx >> 4) & 31);
    long tmp  = idx >> 9;
    int  kt   = (int)(tmp % KT);
    int  nt   = (int)(tmp / KT);
    int  k = kt * 32 + (e >> 3) * 16 + (lane >> 4) * 8 + (e & 7);
    int  n = nt * 16 + (lane & 15);
    float v = (k < Ksrc) ? src[(size_t)k * NG + n] : 0.0f;
    dst[idx] = (bf16_t)v;
}

// ---------------------------------------------------------------------------
// input GEMM: xz = A @ Wx + bias, emitted directly in WMMA C-fragment layout
//   xzp[d][nt][t][lane][r]  (bf16), M-tile = 16 batches at fixed t.
// grid = (T, NG/256, 2 dirs); 8 waves, 2 adjacent N-tiles per wave sharing A.
// ---------------------------------------------------------------------------
__global__ void __launch_bounds__(256)
gemm_in_kernel(const bf16_t* __restrict__ A, int K,
               const bf16_t* __restrict__ Wp, unsigned long wp_dstride,
               const float* __restrict__ bias, int bias_dstride,
               bf16_t* __restrict__ xzp) {
    int t    = blockIdx.x;
    int cg   = blockIdx.y;
    int d    = blockIdx.z;
    int w    = threadIdx.x >> 5;
    int lane = threadIdx.x & 31;
    int nt0  = cg * 16 + w * 2;
    int KT   = K >> 5;

    const bf16_t* wp0 = Wp + (size_t)d * wp_dstride + ((size_t)nt0 * KT) * 512 + lane * 16;
    const bf16_t* wp1 = wp0 + (size_t)KT * 512;
    int b    = lane & 15;                       // batch index = M
    int arow = b * T_ + t;
    const bf16_t* ap = A + (size_t)arow * K + ((lane >> 4) * 8);

    v8f acc0 = {}, acc1 = {};
    for (int kt = 0; kt < KT; ++kt) {
        v16bf a  = ld_afrag(ap + kt * 32);
        v16bf b0 = ld_bfrag(wp0 + kt * 512);
        v16bf b1 = ld_bfrag(wp1 + kt * 512);
        acc0 = __builtin_amdgcn_wmma_f32_16x16x32_bf16(false, a, false, b0,
                                                       (short)0, acc0, false, false);
        acc1 = __builtin_amdgcn_wmma_f32_16x16x32_bf16(false, a, false, b1,
                                                       (short)0, acc1, false, false);
    }
    float bv0 = bias[(size_t)d * bias_dstride + nt0 * 16 + (lane & 15)];
    float bv1 = bias[(size_t)d * bias_dstride + (nt0 + 1) * 16 + (lane & 15)];
    v8bf o0, o1;
#pragma unroll
    for (int r = 0; r < 8; ++r) {
        o0[r] = (bf16_t)(acc0[r] + bv0);
        o1[r] = (bf16_t)(acc1[r] + bv1);
    }
    size_t base0 = (((size_t)(d * 256 + nt0) * T_) + t) * 256 + lane * 8;
    size_t base1 = (((size_t)(d * 256 + nt0 + 1) * T_) + t) * 256 + lane * 8;
    *(v8bf*)(xzp + base0) = o0;
    *(v8bf*)(xzp + base1) = o1;
}

// ---------------------------------------------------------------------------
// grid barrier across NWG_DIR workgroups of one direction (phase barrier,
// self-resetting so it is consistent across graph replays)
// ---------------------------------------------------------------------------
__device__ __forceinline__ void grid_barrier(unsigned* cnt, unsigned* phase) {
    __syncthreads();
    if (threadIdx.x == 0) {
        __threadfence();
        unsigned old  = __hip_atomic_load(phase, __ATOMIC_RELAXED, __HIP_MEMORY_SCOPE_AGENT);
        unsigned prev = __hip_atomic_fetch_add(cnt, 1u, __ATOMIC_ACQ_REL, __HIP_MEMORY_SCOPE_AGENT);
        if (prev == NWG_DIR - 1u) {
            __hip_atomic_store(cnt, 0u, __ATOMIC_RELAXED, __HIP_MEMORY_SCOPE_AGENT);
            __hip_atomic_fetch_add(phase, 1u, __ATOMIC_RELEASE, __HIP_MEMORY_SCOPE_AGENT);
        } else {
            while (__hip_atomic_load(phase, __ATOMIC_ACQUIRE, __HIP_MEMORY_SCOPE_AGENT) == old)
                __builtin_amdgcn_s_sleep(1);
        }
    }
    __syncthreads();
}

// ---------------------------------------------------------------------------
// persistent recurrent kernel: one layer, both directions.
// grid = 64 WGs: blockIdx>>5 = dir, &31 = hidden-unit slice (32 units).
// wave w handles gate g=w>>1, sub-tile w&1 -> col tile nt = g*64 + wg*2 + (w&1)
// ---------------------------------------------------------------------------
__global__ void __launch_bounds__(256)
lstm_rec_kernel(const bf16_t* __restrict__ xzp,    // [2][256][T][256] C-frag bf16
                const bf16_t* __restrict__ whp,    // [2][256][32][32][16] packed
                const int*    __restrict__ xmask,  // [B][T]
                bf16_t*       __restrict__ hout,   // [ROWS][2048]
                bf16_t*       __restrict__ hstate, // [2][2][16][1024] ping-pong
                unsigned*     __restrict__ bar) {  // [2][2] cnt/phase per dir
    __shared__ bf16_t hsh[16 * UNITS_];  // 32 KB: h(t-1) for whole batch
    __shared__ float  zsh[8][16][16];    //  8 KB: z tiles for gate assembly
    __shared__ float  csh[16][32];       //  2 KB: cell state (persistent)

    int d    = blockIdx.x >> 5;
    int wg   = blockIdx.x & 31;
    int tid  = threadIdx.x;
    int w    = tid >> 5, lane = tid & 31;
    int u0   = wg * 32;
    int g    = w >> 1;
    int nt   = g * 64 + wg * 2 + (w & 1);

    const bf16_t* wpb = whp + (size_t)d * (256u * 32u * 512u)
                            + (size_t)nt * (32u * 512u) + lane * 16;
    const bf16_t* xzf = xzp + ((size_t)(d * 256 + nt) * T_) * 256 + lane * 8;
    bf16_t*   hs_base = hstate + (size_t)d * (2 * 16 * UNITS_);
    unsigned* cnt = bar + d * 2;
    unsigned* phs = cnt + 1;

    // zero cell state and this WG's slice of h-state slot 0
    for (int i = tid; i < 16 * 32; i += 256) ((float*)csh)[i] = 0.0f;
    for (int i = tid; i < 512; i += 256) {
        int m = i >> 5, u = i & 31;
        hs_base[m * UNITS_ + u0 + u] = (bf16_t)0.0f;
    }
    grid_barrier(cnt, phs);

    int mb = (lane >> 4) * 8;
    const bf16_t* ash = &hsh[(lane & 15) * UNITS_ + (lane >> 4) * 8];

    for (int s = 0; s < T_; ++s) {
        int t = d ? (T_ - 1 - s) : s;

        // stage h(t-1) into LDS (16x1024 bf16 = 2048 uint4)
        const uint4* hrd = (const uint4*)(hs_base + (size_t)(s & 1) * 16 * UNITS_);
        for (int i = tid; i < 2048; i += 256) ((uint4*)hsh)[i] = hrd[i];
        __syncthreads();

        // accumulator init: one b128 load of the pre-packed xz C-fragment
        uint4 xv = *(const uint4*)(xzf + (size_t)t * 256);
        v8bf  xb = __builtin_bit_cast(v8bf, xv);
        v8f acc;
#pragma unroll
        for (int r = 0; r < 8; ++r) acc[r] = (float)xb[r];

        // z += h @ Wh  (per-wave 16x16 tile, K=1024)
#pragma unroll 4
        for (int kt = 0; kt < 32; ++kt) {
            v16bf a = ld_afrag(ash + kt * 32);     // LDS
            v16bf b = ld_bfrag(wpb + kt * 512);    // L2-resident weights
            acc = __builtin_amdgcn_wmma_f32_16x16x32_bf16(false, a, false, b,
                                                          (short)0, acc, false, false);
        }
#pragma unroll
        for (int r = 0; r < 8; ++r) zsh[w][mb + r][lane & 15] = acc[r];
        __syncthreads();

        // gates: 16 batch rows x 32 units = 512 elements, 2 per thread
#pragma unroll
        for (int rep = 0; rep < 2; ++rep) {
            int idx = tid + rep * 256;
            int m = idx >> 5, u = idx & 31;
            int st = u >> 4, n = u & 15;
            float zi = zsh[0 + st][m][n];
            float zf = zsh[2 + st][m][n];
            float zg = zsh[4 + st][m][n];
            float zo = zsh[6 + st][m][n];
            float co = csh[m][u];
            float cn = sigf(zf) * co + sigf(zi) * tanhf(zg);
            float hn = sigf(zo) * tanhf(cn);
            bool  valid = (xmask[m * T_ + t] != 1);
            float hprev = (float)hsh[m * UNITS_ + u0 + u];
            float hv = valid ? hn : hprev;
            float cv = valid ? cn : co;
            csh[m][u] = cv;
            bf16_t hb = (bf16_t)hv;
            hs_base[((s + 1) & 1) * 16 * UNITS_ + m * UNITS_ + u0 + u] = hb;
            hout[(size_t)(m * T_ + t) * (2 * UNITS_) + d * UNITS_ + u0 + u] = hb;
        }
        grid_barrier(cnt, phs);  // h(t) fully published before next step
    }
}

// ---------------------------------------------------------------------------
// heads: relu(h) @ {W8,W3,Wpp} + bias -> concat fp32 output
// ---------------------------------------------------------------------------
__global__ void head_kernel(const bf16_t* __restrict__ h,
                            const float* __restrict__ W8,  const float* __restrict__ b8,
                            const float* __restrict__ W3,  const float* __restrict__ b3,
                            const float* __restrict__ Wpp, const float* __restrict__ bpp,
                            float* __restrict__ out) {
    int idx = blockIdx.x * 256 + threadIdx.x;
    int row = idx >> 4, s = idx & 15;
    if (row >= ROWS || s >= 15) return;
    const float *W, *bb; int nc, c; float* o;
    if (s < 8)       { W = W8;  bb = b8;  nc = 8; c = s;      o = out; }
    else if (s < 11) { W = W3;  bb = b3;  nc = 3; c = s - 8;  o = out + (size_t)ROWS * 8; }
    else             { W = Wpp; bb = bpp; nc = 4; c = s - 11; o = out + (size_t)ROWS * 11; }
    const bf16_t* hr = h + (size_t)row * (2 * UNITS_);
    float acc = bb[c];
    for (int k = 0; k < 2 * UNITS_; ++k) {
        float hv = (float)hr[k];
        hv = hv > 0.0f ? hv : 0.0f;
        acc += hv * W[k * nc + c];
    }
    o[(size_t)row * nc + c] = acc;
}

// ---------------------------------------------------------------------------
extern "C" void kernel_launch(void* const* d_in, const int* in_sizes, int n_in,
                              void* d_out, int out_size, void* d_ws, size_t ws_size,
                              hipStream_t stream) {
    (void)in_sizes; (void)n_in; (void)out_size; (void)ws_size;
    const float* x   = (const float*)d_in[0];
    const int*   xm  = (const int*)d_in[1];
    const float* Wx0 = (const float*)d_in[3];
    const float* Wh0 = (const float*)d_in[4];
    const float* b0  = (const float*)d_in[5];
    const float* WxL = (const float*)d_in[6];
    const float* WhL = (const float*)d_in[7];
    const float* bL  = (const float*)d_in[8];
    const float* W8  = (const float*)d_in[9];
    const float* b8  = (const float*)d_in[10];
    const float* W3  = (const float*)d_in[11];
    const float* b3  = (const float*)d_in[12];
    const float* Wpp = (const float*)d_in[13];
    const float* bpp = (const float*)d_in[14];
    float* out = (float*)d_out;

    char* ws = (char*)d_ws;
    size_t off = 0;
    auto take = [&](size_t bytes) -> char* {
        char* p = ws + off;
        off = (off + bytes + 255) & ~(size_t)255;
        return p;
    };
    bf16_t*   xbf    = (bf16_t*)take((size_t)ROWS * K0P * 2);
    bf16_t*   wxp0   = (bf16_t*)take((size_t)2 * K0P * NG * 2);
    bf16_t*   wxpL   = (bf16_t*)take((size_t)3 * 2 * 2048 * NG * 2);
    bf16_t*   whp    = (bf16_t*)take((size_t)4 * 2 * UNITS_ * NG * 2);
    bf16_t*   xzp    = (bf16_t*)take((size_t)2 * ROWS * NG * 2);
    bf16_t*   hA     = (bf16_t*)take((size_t)ROWS * 2 * UNITS_ * 2);
    bf16_t*   hB     = (bf16_t*)take((size_t)ROWS * 2 * UNITS_ * 2);
    bf16_t*   hstate = (bf16_t*)take((size_t)2 * 2 * 16 * UNITS_ * 2);
    unsigned* bar    = (unsigned*)take(256);

    init_kernel<<<1, 256, 0, stream>>>(hstate, bar);
    convx_kernel<<<(ROWS * K0P + 255) / 256, 256, 0, stream>>>(x, xbf);

    // pack weights to WMMA B layout (bf16)
    for (int d = 0; d < 2; ++d) {
        long tot = (long)K0P * NG;
        pack_b_kernel<<<(unsigned)((tot + 255) / 256), 256, 0, stream>>>(
            Wx0 + (size_t)d * FEAT_ * NG, wxp0 + (size_t)d * K0P * NG, FEAT_, K0P / 32, tot);
    }
    for (int l = 0; l < 3; ++l)
        for (int d = 0; d < 2; ++d) {
            long tot = (long)2048 * NG;
            pack_b_kernel<<<(unsigned)((tot + 255) / 256), 256, 0, stream>>>(
                WxL + ((size_t)(l * 2 + d)) * 2048 * NG,
                wxpL + ((size_t)(l * 2 + d)) * 2048 * NG, 2048, 64, tot);
        }
    for (int d = 0; d < 2; ++d) {
        long tot = (long)UNITS_ * NG;
        pack_b_kernel<<<(unsigned)((tot + 255) / 256), 256, 0, stream>>>(
            Wh0 + (size_t)d * UNITS_ * NG, whp + (size_t)d * UNITS_ * NG, UNITS_, 32, tot);
        for (int l = 0; l < 3; ++l)
            pack_b_kernel<<<(unsigned)((tot + 255) / 256), 256, 0, stream>>>(
                WhL + ((size_t)(l * 2 + d)) * UNITS_ * NG,
                whp + ((size_t)((l + 1) * 2 + d)) * UNITS_ * NG, UNITS_, 32, tot);
    }

    // layers
    const bf16_t* inA = xbf;
    int K = K0P;
    for (int l = 0; l < 4; ++l) {
        const bf16_t* wx = (l == 0) ? wxp0 : wxpL + (size_t)(l - 1) * 2 * 2048 * NG;
        unsigned long wx_dstride = (l == 0) ? (size_t)K0P * NG : (size_t)2048 * NG;
        const float* bias = (l == 0) ? b0 : bL + (size_t)(l - 1) * 2 * NG;
        gemm_in_kernel<<<dim3(T_, NG / 256, 2), 256, 0, stream>>>(
            inA, K, wx, wx_dstride, bias, NG, xzp);
        bf16_t* outh = (l & 1) ? hB : hA;
        lstm_rec_kernel<<<2 * NWG_DIR, 256, 0, stream>>>(
            xzp, whp + (size_t)l * 2 * UNITS_ * NG, xm, outh, hstate, bar);
        inA = outh;
        K = 2 * UNITS_;
    }

    head_kernel<<<(ROWS * 16 + 255) / 256, 256, 0, stream>>>(
        hB, W8, b8, W3, b3, Wpp, bpp, out);
}
